// CrossAttention_24601572671942
// MI455X (gfx1250) — matrix operand order repaired and checked by reading, not compiled, not run on hardware
//
#include <hip/hip_runtime.h>
#include <stdint.h>

// ---------------------------------------------------------------------------
// CrossAttention forward for MI455X (gfx1250), bf16 WMMA path, flash softmax.
// B=4, T=1024, C=1024, H=16, DH=64.
// ---------------------------------------------------------------------------

typedef __bf16 bf16;
typedef __attribute__((ext_vector_type(16))) __bf16 v16bf;
typedef __attribute__((ext_vector_type(8)))  float  v8f;
typedef int v4i __attribute__((__vector_size__(4 * sizeof(int))));

union FragU { v16bf v; uint4 q[2]; };

__device__ __forceinline__ v8f wmma_bf16(v16bf a, v16bf b, v8f c) {
  return __builtin_amdgcn_wmma_f32_16x16x32_bf16(
      /*neg_a=*/false, a, /*neg_b=*/false, b,
      /*c_mod=*/(short)0, c, /*reuse_a=*/false, /*reuse_b=*/false);
}

// 16B global -> LDS copy. Prefer the CDNA5 async path (GLOBAL_LOAD_ASYNC_TO_
// LDS_B128, ASYNCcnt) which bypasses VGPRs; fall back to load+ds_store.
// Builtin signature (from the round-2 diagnostic): param 0 is
// 'v4i __device__*' (addrspace(1) pointer to int4). Flat shared addresses
// carry the LDS offset in the low 32 bits (aperture truncation) and global
// VAs are identical in as(1), so the integer round-trip casts are exact.
#if __has_builtin(__builtin_amdgcn_global_load_async_to_lds_b128)
#define ASYNC_LDS_COPY 1
#else
#define ASYNC_LDS_COPY 0
#endif

__device__ __forceinline__ void copy16_g2l(const bf16* g, bf16* l) {
#if ASYNC_LDS_COPY
  __builtin_amdgcn_global_load_async_to_lds_b128(
      (__attribute__((address_space(1))) v4i*)(uintptr_t)g,
      (__attribute__((address_space(3))) v4i*)(uintptr_t)l,
      0, 0);
#else
  *(uint4*)l = *(const uint4*)g;
#endif
}

__device__ __forceinline__ void copy_wait() {
#if ASYNC_LDS_COPY
  asm volatile("s_wait_asynccnt 0" ::: "memory");
#endif
}

// A-fragment (16x32, 16-bit): lane holds row (lane&15); K chunks
// {base..base+7} and {base+16..base+23}, base = (lane>>4)*8.  (ISA 7.12.2)
__device__ __forceinline__ v16bf load_frag_A(const bf16* p, int row, int stride,
                                             int kbase, int lane) {
  FragU f;
  int c0 = kbase + ((lane >> 4) << 3);
  f.q[0] = *(const uint4*)(p + row * stride + c0);
  f.q[1] = *(const uint4*)(p + row * stride + c0 + 16);
  return f.v;
}

// B-fragment (32x16, 16-bit): lane holds column (lane&15); K chunks
// {base..base+7} and {base+16..base+23}, base = (lane>>4)*8.  Memory is B^T row-major (K contiguous).
__device__ __forceinline__ v16bf load_frag_B(const bf16* p, int row, int stride,
                                             int kbase, int lane) {
  FragU f;
  int c0 = kbase + ((lane >> 4) << 3);
  f.q[0] = *(const uint4*)(p + row * stride + c0);
  f.q[1] = *(const uint4*)(p + row * stride + c0 + 16);
  return f.v;
}

// ---------------------------------------------------------------------------
// fp32 -> bf16 conversion, float4-vectorized
// ---------------------------------------------------------------------------
__global__ __launch_bounds__(256) void cvt_f32_bf16(const float* __restrict__ in,
                                                    bf16* __restrict__ out, int n4) {
  int i = blockIdx.x * 256 + threadIdx.x;
  if (i < n4) {
    float4 f = ((const float4*)in)[i];
    bf16 o[4] = {(bf16)f.x, (bf16)f.y, (bf16)f.z, (bf16)f.w};
    *(ushort4*)(out + 4 * (size_t)i) = *(const ushort4*)o;
  }
}

// ---------------------------------------------------------------------------
// GEMM: D[m,n] = sum_k A[m,k] * W[n,k]   (M=4096, N=1024, K=1024)
// mode 0: Q -> [B,H,T,DH] bf16, alpha=DH^-0.5
// mode 1: K -> [B,H,T,DH] bf16
// mode 2: V -> [B,H,DH,T] bf16 (transposed for P@V B-fragments)
// mode 3: out = D + bias -> fp32 [M,N]
// Block 256 thr = 8 waves (4x2), block tile 128x128, wave tile 32x64,
// K staged 64-deep: 16 WMMA per barrier pair, 1.5 ds_load_b128 per WMMA.
// ---------------------------------------------------------------------------
__global__ __launch_bounds__(256) void gemm_bf16_k(
    const bf16* __restrict__ A, const bf16* __restrict__ W,
    bf16* __restrict__ obf, float* __restrict__ of32,
    const float* __restrict__ bias, int mode, float alpha) {
  __shared__ bf16 As[128 * 72];  // 64-wide K stage, pad 8 -> 144B row stride
  __shared__ bf16 Bs[128 * 72];

  const int K = 1024;
  int tid  = threadIdx.x;
  int lane = tid & 31, wave = tid >> 5;
  int m0 = blockIdx.x * 128, n0 = blockIdx.y * 128;
  int wm = (wave >> 1) * 32, wn = (wave & 1) * 64;

  int sr = tid >> 2;            // staging row 0..63 (two passes: sr, sr+64)
  int sc = (tid & 3) * 16;      // element offset within 64-wide K stage

  v8f acc[2][4] = {};

  for (int k0 = 0; k0 < K; k0 += 64) {
    copy16_g2l(A + (size_t)(m0 + sr)      * K + k0 + sc,     &As[sr * 72 + sc]);
    copy16_g2l(A + (size_t)(m0 + sr)      * K + k0 + sc + 8, &As[sr * 72 + sc + 8]);
    copy16_g2l(A + (size_t)(m0 + sr + 64) * K + k0 + sc,     &As[(sr + 64) * 72 + sc]);
    copy16_g2l(A + (size_t)(m0 + sr + 64) * K + k0 + sc + 8, &As[(sr + 64) * 72 + sc + 8]);
    copy16_g2l(W + (size_t)(n0 + sr)      * K + k0 + sc,     &Bs[sr * 72 + sc]);
    copy16_g2l(W + (size_t)(n0 + sr)      * K + k0 + sc + 8, &Bs[sr * 72 + sc + 8]);
    copy16_g2l(W + (size_t)(n0 + sr + 64) * K + k0 + sc,     &Bs[(sr + 64) * 72 + sc]);
    copy16_g2l(W + (size_t)(n0 + sr + 64) * K + k0 + sc + 8, &Bs[(sr + 64) * 72 + sc + 8]);
    if (k0 + 64 < K) {  // pull the next K stage toward this WGP
      __builtin_prefetch(A + (size_t)(m0 + sr) * K + k0 + 64 + sc, 0, 3);
      __builtin_prefetch(W + (size_t)(n0 + sr) * K + k0 + 64 + sc, 0, 3);
    }
    copy_wait();
    __syncthreads();

#pragma unroll
    for (int kf = 0; kf < 2; kf++) {
      v16bf af[2], bw[4];
#pragma unroll
      for (int i = 0; i < 2; i++)
        af[i] = load_frag_A(As, wm + i * 16 + (lane & 15), 72, kf * 32, lane);
#pragma unroll
      for (int j = 0; j < 4; j++)
        bw[j] = load_frag_B(Bs, wn + j * 16 + (lane & 15), 72, kf * 32, lane);
#pragma unroll
      for (int i = 0; i < 2; i++)
#pragma unroll
        for (int j = 0; j < 4; j++) acc[i][j] = wmma_bf16(af[i], bw[j], acc[i][j]);
    }
    __syncthreads();
  }

  // Epilogue. C layout: row = r + 8*(lane>>4), col = lane&15.
#pragma unroll
  for (int i = 0; i < 2; i++)
#pragma unroll
    for (int j = 0; j < 4; j++) {
      int nc = n0 + wn + j * 16 + (lane & 15);
      float bv = (mode == 3) ? bias[nc] : 0.0f;
#pragma unroll
      for (int r = 0; r < 8; r++) {
        int mr = m0 + wm + i * 16 + ((lane >> 4) << 3) + r;
        float v = acc[i][j][r] * alpha;
        if (mode == 3) {
          of32[(size_t)mr * 1024 + nc] = v + bv;
        } else {
          int b = mr >> 10, t = mr & 1023;
          int h = nc >> 6,  d = nc & 63;
          size_t idx = (mode == 2)
              ? ((size_t)(b * 16 + h) * 64 + d) * 1024 + t     // V^T: [B,H,DH,T]
              : ((size_t)(b * 16 + h) * 1024 + t) * 64 + d;    // Q/K: [B,H,T,DH]
          obf[idx] = (bf16)v;
        }
      }
    }
}

// ---------------------------------------------------------------------------
// Flash attention: one wave owns 16 query rows, block = 8 waves = 128 rows.
// Key blocks of 64 staged in LDS (shared by all waves). Running softmax.
// grid = (B*H, T/128)
// ---------------------------------------------------------------------------
__global__ __launch_bounds__(256) void flash_attn_k(
    const bf16* __restrict__ Q, const bf16* __restrict__ Kb,
    const bf16* __restrict__ Vt, const unsigned char* __restrict__ mask,
    bf16* __restrict__ Y) {
  __shared__ bf16 Ks[64 * 72];       // K rows j..j+63, DH=64 + pad
  __shared__ bf16 Vs[64 * 72];       // V^T rows d=0..63, key cols j..j+63 + pad
  __shared__ bf16 Ps[8][16 * 72];    // per-wave P staging (C-layout -> A-frag)

  const int T = 1024, DH = 64;
  int tid = threadIdx.x, lane = tid & 31, wave = tid >> 5;
  int bh = blockIdx.x;
  int b = bh >> 4, h = bh & 15;
  int i0 = blockIdx.y * 128 + wave * 16;

  const bf16* Qp = Q  + ((size_t)bh * T + i0) * DH;
  const bf16* Kp = Kb + (size_t)bh * T * DH;
  const bf16* Vp = Vt + (size_t)bh * DH * T;
  const unsigned char* Mp = mask + (size_t)b * T * T;

  // Q fragments for this wave's 16 rows (K-dim = DH = 64 -> 2 frags)
  v16bf aq[2];
#pragma unroll
  for (int kf = 0; kf < 2; kf++) {
    FragU f;
    int row = lane & 15;
    int c0 = kf * 32 + ((lane >> 4) << 3);
    f.q[0] = *(const uint4*)(Qp + row * DH + c0);
    f.q[1] = *(const uint4*)(Qp + row * DH + c0 + 16);
    aq[kf] = f.v;
  }

  float mrow[8], rsum[8];
  v8f o[4] = {};
#pragma unroll
  for (int r = 0; r < 8; r++) { mrow[r] = -3.402823466e38f; rsum[r] = 0.0f; }

  int sr = tid >> 2;            // staging row 0..63
  int sc = (tid & 3) * 16;      // 16 elements (32 B) per thread per row

  for (int j0 = 0; j0 < T; j0 += 64) {
    copy16_g2l(Kp + (size_t)(j0 + sr) * DH + sc,     &Ks[sr * 72 + sc]);
    copy16_g2l(Kp + (size_t)(j0 + sr) * DH + sc + 8, &Ks[sr * 72 + sc + 8]);
    copy16_g2l(Vp + (size_t)sr * T + j0 + sc,        &Vs[sr * 72 + sc]);
    copy16_g2l(Vp + (size_t)sr * T + j0 + sc + 8,    &Vs[sr * 72 + sc + 8]);
    copy_wait();
    __syncthreads();

    // S = Q @ K^T for 16x64 tile (4 N-tiles x 2 K-steps)
    v8f st[4];
#pragma unroll
    for (int nt = 0; nt < 4; nt++) {
      v8f s = {};
#pragma unroll
      for (int kf = 0; kf < 2; kf++) {
        v16bf bk = load_frag_B(Ks, nt * 16 + (lane & 15), 72, kf * 32, lane);
        s = wmma_bf16(aq[kf], bk, s);
      }
      st[nt] = s;
    }

    // mask (bool [B,1,T,T]) applied in C-layout
    int ibase = i0 + ((lane >> 4) << 3);
#pragma unroll
    for (int nt = 0; nt < 4; nt++) {
      int j = j0 + nt * 16 + (lane & 15);
#pragma unroll
      for (int r = 0; r < 8; r++) {
        unsigned char mv = Mp[(size_t)(ibase + r) * T + j];
        st[nt][r] = mv ? st[nt][r] : -3.402823466e38f;
      }
    }

    // running softmax: row lives across 16 lanes of one half-wave
#pragma unroll
    for (int r = 0; r < 8; r++) {
      float tm = fmaxf(fmaxf(st[0][r], st[1][r]), fmaxf(st[2][r], st[3][r]));
#pragma unroll
      for (int off = 1; off < 16; off <<= 1) tm = fmaxf(tm, __shfl_xor(tm, off, 32));
      float mn = fmaxf(mrow[r], tm);
      float corr = __expf(mrow[r] - mn);
      float ls = 0.0f;
#pragma unroll
      for (int nt = 0; nt < 4; nt++) {
        float p = __expf(st[nt][r] - mn);
        st[nt][r] = p;
        ls += p;
      }
#pragma unroll
      for (int off = 1; off < 16; off <<= 1) ls += __shfl_xor(ls, off, 32);
      rsum[r] = rsum[r] * corr + ls;
      mrow[r] = mn;
#pragma unroll
      for (int nt = 0; nt < 4; nt++) o[nt][r] *= corr;
    }

    // transpose P through LDS: C-layout -> A-fragment layout
    bf16* Pw = Ps[wave];
#pragma unroll
    for (int nt = 0; nt < 4; nt++)
#pragma unroll
      for (int r = 0; r < 8; r++)
        Pw[(((lane >> 4) << 3) + r) * 72 + nt * 16 + (lane & 15)] = (bf16)st[nt][r];
    asm volatile("s_wait_dscnt 0" ::: "memory");  // P stores -> P frag loads (same wave)

    // O += P @ V   (K-dim = key block 64 -> 2 frags; N = DH -> 4 tiles)
#pragma unroll
    for (int kf = 0; kf < 2; kf++) {
      v16bf ap = load_frag_A(Pw, lane & 15, 72, kf * 32, lane);
#pragma unroll
      for (int dt = 0; dt < 4; dt++) {
        v16bf bv = load_frag_B(Vs, dt * 16 + (lane & 15), 72, kf * 32, lane);
        o[dt] = wmma_bf16(ap, bv, o[dt]);
      }
    }
    __syncthreads();
  }

  // normalize and store Y [B,T,C] bf16 (heads merged)
#pragma unroll
  for (int r = 0; r < 8; r++) {
    float inv = 1.0f / rsum[r];
    int i = i0 + ((lane >> 4) << 3) + r;
#pragma unroll
    for (int dt = 0; dt < 4; dt++) {
      int d = h * 64 + dt * 16 + (lane & 15);
      Y[(size_t)(b * T + i) * 1024 + d] = (bf16)(o[dt][r] * inv);
    }
  }
}

// ---------------------------------------------------------------------------
// launch
// ---------------------------------------------------------------------------
extern "C" void kernel_launch(void* const* d_in, const int* in_sizes, int n_in,
                              void* d_out, int out_size, void* d_ws, size_t ws_size,
                              hipStream_t stream) {
  (void)in_sizes; (void)n_in; (void)out_size; (void)ws_size;
  const float* x  = (const float*)d_in[0];
  // d_in[1] = context (dead in reference forward)
  const unsigned char* mask = (const unsigned char*)d_in[2];  // bool [B,1,T,T]
  const float* Wq = (const float*)d_in[3];
  const float* Wk = (const float*)d_in[4];
  const float* Wv = (const float*)d_in[5];
  const float* Wo = (const float*)d_in[6];
  const float* bo = (const float*)d_in[7];
  float* out = (float*)d_out;

  char* ws = (char*)d_ws;
  const size_t MB = 1u << 20;
  bf16* xbf = (bf16*)(ws + 0 * MB);    // 4096x1024        (8 MiB)
  bf16* wqb = (bf16*)(ws + 8 * MB);    // 1024x1024        (2 MiB)
  bf16* wkb = (bf16*)(ws + 10 * MB);
  bf16* wvb = (bf16*)(ws + 12 * MB);
  bf16* wob = (bf16*)(ws + 14 * MB);
  bf16* Qb  = (bf16*)(ws + 16 * MB);   // [B,H,T,DH]       (8 MiB)
  bf16* Kb  = (bf16*)(ws + 24 * MB);   // [B,H,T,DH]
  bf16* Vtb = (bf16*)(ws + 32 * MB);   // [B,H,DH,T]
  bf16* Yb  = (bf16*)(ws + 40 * MB);   // [B,T,C]

  // fp32 -> bf16 conversions (one pass; halves all GEMM operand traffic)
  cvt_f32_bf16<<<4096, 256, 0, stream>>>(x,  xbf, 4096 * 1024 / 4);
  cvt_f32_bf16<<<1024, 256, 0, stream>>>(Wq, wqb, 1024 * 1024 / 4);
  cvt_f32_bf16<<<1024, 256, 0, stream>>>(Wk, wkb, 1024 * 1024 / 4);
  cvt_f32_bf16<<<1024, 256, 0, stream>>>(Wv, wvb, 1024 * 1024 / 4);
  cvt_f32_bf16<<<1024, 256, 0, stream>>>(Wo, wob, 1024 * 1024 / 4);

  dim3 gg(32, 8);  // M/128 x N/128
  gemm_bf16_k<<<gg, 256, 0, stream>>>(xbf, wqb, Qb,  nullptr, nullptr, 0, 0.125f); // Q*DH^-0.5
  gemm_bf16_k<<<gg, 256, 0, stream>>>(xbf, wkb, Kb,  nullptr, nullptr, 1, 1.0f);
  gemm_bf16_k<<<gg, 256, 0, stream>>>(xbf, wvb, Vtb, nullptr, nullptr, 2, 1.0f);

  flash_attn_k<<<dim3(64, 8), 256, 0, stream>>>(Qb, Kb, Vtb, mask, Yb);

  gemm_bf16_k<<<gg, 256, 0, stream>>>(Yb, wob, nullptr, out, bo, 3, 1.0f);
}
